// MultiHeadAttention_73555609911368
// MI455X (gfx1250) — matrix operand order repaired and checked by reading, not compiled, not run on hardware
//
#include <hip/hip_runtime.h>

typedef __bf16 bf16_t;
typedef __attribute__((ext_vector_type(16))) __bf16 v16bf;
typedef __attribute__((ext_vector_type(8)))  __bf16 v8bf;
typedef __attribute__((ext_vector_type(8)))  float  v8f;

#define EMB   1024
#define NHEAD 16
#define HDIM  64
#define SEQ   2048
#define BATCH 4
#define NTOK  (BATCH*SEQ)   /* 8192 */
#define N3    (3*EMB)       /* 3072 */

// CDNA5 async global->LDS staging (ASYNCcnt path); proven to assemble.
#define USE_ASYNC_LDS 1

static __device__ __forceinline__ void stage16(bf16_t* lds, const bf16_t* g) {
#if USE_ASYNC_LDS
  unsigned off = (unsigned)(size_t)lds;          // low 32 bits = LDS offset
  asm volatile("global_load_async_to_lds_b128 %0, %1, off"
               :: "v"(off), "v"(g) : "memory");
#else
  *(v8bf*)lds = *(const v8bf*)g;
#endif
}
static __device__ __forceinline__ void stage_wait() {
#if USE_ASYNC_LDS
  asm volatile("s_wait_asynccnt 0x0" ::: "memory");
#endif
}

static __device__ __forceinline__ v16bf make_a_frag(v8bf lo, v8bf hi) {
  v16bf a;
#pragma unroll
  for (int i = 0; i < 8; ++i) { a[i] = lo[i]; a[8 + i] = hi[i]; }
  return a;
}

static __device__ __forceinline__ v8f wmma_bf16(v16bf a, v16bf b, v8f c) {
  return __builtin_amdgcn_wmma_f32_16x16x32_bf16(false, a, false, b,
                                                 (short)0, c, false, false);
}

// ---------------------------------------------------------------------------
// Pack fp32 row-major [K x N] weights into bf16 WMMA B-fragment order
// [kt][nt][lane][e]: one contiguous 32B fragment per lane per WMMA.
// ---------------------------------------------------------------------------
__global__ void __launch_bounds__(256)
pack_w_kernel(const float* __restrict__ W, bf16_t* __restrict__ Wp, int N) {
  int idx    = blockIdx.x * 256 + threadIdx.x;
  int e      = idx & 15;
  int lane   = (idx >> 4) & 31;
  int frag   = idx >> 9;
  int ntiles = N >> 4;
  int kt     = frag / ntiles;
  int nt     = frag - kt * ntiles;
  int k      = kt * 32 + ((e & 8) ? e + 8 : e) + ((lane >> 4) << 3);
  int n      = nt * 16 + (lane & 15);
  Wp[idx] = (bf16_t)W[(size_t)k * N + n];
}

// ---------------------------------------------------------------------------
// Pack fp32 x [8192 x 1024] into bf16 WMMA A-fragment order [mt][kt][lane][e].
// ---------------------------------------------------------------------------
__global__ void __launch_bounds__(256)
pack_x_kernel(const float* __restrict__ x, bf16_t* __restrict__ Xp) {
  int idx  = blockIdx.x * 256 + threadIdx.x;
  int e    = idx & 15;
  int lane = (idx >> 4) & 31;
  int frag = idx >> 9;          // mt*32 + kt
  int kt   = frag & 31;
  int mt   = frag >> 5;
  int m    = mt * 16 + (lane & 15);
  int k    = kt * 32 + ((e & 8) ? e + 8 : e) + ((lane & 16) ? 8 : 0);
  Xp[idx] = (bf16_t)x[(size_t)m * EMB + k];
}

// ---------------------------------------------------------------------------
// QKV projection. Block = 8 m-tiles x 1 n-group. Double-buffered 8 KB LDS
// B-panels: async-stage panel k+1 while 64 WMMAs consume panel k; one
// barrier per k-step. B fragments preloaded as a register array so the
// dscnt waits pipeline instead of stalling each WMMA.
// ---------------------------------------------------------------------------
__global__ void __launch_bounds__(256)
qkv_kernel(const bf16_t* __restrict__ Xp, const bf16_t* __restrict__ Wp,
           const float* __restrict__ bias,
           bf16_t* __restrict__ Q, bf16_t* __restrict__ Kk,
           bf16_t* __restrict__ Vt) {
  const int NT = N3 / 16;                         // 192 column tiles
  const int NG = NT / 8;                          // 24 n-groups
  __shared__ __attribute__((aligned(32))) bf16_t lbs[2][8 * 512]; // 16 KB

  int mg   = blockIdx.x / NG;
  int ng   = blockIdx.x - mg * NG;
  int wib  = threadIdx.x >> 5;
  int lane = threadIdx.x & 31;
  int mt   = mg * 8 + wib;
  bool hi  = lane >= 16;
  int  l15 = lane & 15;
  int  tid = threadIdx.x;

  v8f acc[8] = {};
  const bf16_t* ap  = Xp + ((size_t)mt * 32 * 32 + lane) * 16;
  const bf16_t* bkt = Wp + ((size_t)ng * 8) * 512 + tid * 16;

  // prologue: stage panel 0, load A(0)
  stage16(&lbs[0][tid * 16],     bkt);
  stage16(&lbs[0][tid * 16 + 8], bkt + 8);
  bkt += (size_t)NT * 512;
  v16bf a = *(const v16bf*)ap;
  stage_wait();
  __syncthreads();

  for (int kt = 0; kt < EMB / 32; ++kt) {
    const bf16_t* cur = lbs[kt & 1];
    if (kt + 1 < EMB / 32) {                      // prefetch next panel
      bf16_t* nxt = lbs[(kt + 1) & 1];
      stage16(nxt + tid * 16,     bkt);
      stage16(nxt + tid * 16 + 8, bkt + 8);
      bkt += (size_t)NT * 512;
    }
    const bf16_t* apn = ap + ((kt + 1 < EMB / 32) ? 512 : 0);
    v16bf a_next = *(const v16bf*)apn;            // prefetch next A
    ap = apn;

    v16bf bfr[8];                                 // preload -> pipelined waits
#pragma unroll
    for (int t = 0; t < 8; ++t)
      bfr[t] = *(const v16bf*)(cur + lane * 16 + t * 512);
#pragma unroll
    for (int t = 0; t < 8; ++t)
      acc[t] = wmma_bf16(a, bfr[t], acc[t]);
    a = a_next;

    stage_wait();
    __syncthreads();
  }

  int tokbase = mt * 16 + (hi ? 8 : 0);
  int b_      = (mt * 16) >> 11;
#pragma unroll
  for (int t = 0; t < 8; ++t) {
    int n     = ng * 128 + t * 16 + l15;
    float bv  = bias[n];
    int which = n >> 10;                          // 0=Q,1=K,2=V (uniform/tile)
    int c     = n & (EMB - 1);
    int h     = c >> 6;
    int dd    = c & (HDIM - 1);
    int bh    = b_ * NHEAD + h;
#pragma unroll
    for (int r = 0; r < 8; ++r) {
      int t_  = (tokbase + r) & (SEQ - 1);
      float v = acc[t][r] + bv;
      if (which == 0) {
        Q[((size_t)bh * SEQ + t_) * HDIM + dd] = (bf16_t)(v * 0.125f);
      } else if (which == 1) {
        Kk[((size_t)bh * SEQ + t_) * HDIM + dd] = (bf16_t)v;
      } else {
        Vt[((size_t)bh * HDIM + dd) * SEQ + t_] = (bf16_t)v;  // d-major
      }
    }
  }
}

// ---------------------------------------------------------------------------
// Flash attention. Block = 8 q-tiles of one (b,h). Double-buffered async
// staging of the shared 32-key K (32x64) / V (64x32 d-major) tiles; one
// barrier per chunk. K and V fragments preloaded before their WMMA chains.
// ---------------------------------------------------------------------------
__global__ void __launch_bounds__(256)
attn_kernel(const bf16_t* __restrict__ Q, const bf16_t* __restrict__ Kk,
            const bf16_t* __restrict__ Vt, bf16_t* __restrict__ Ao) {
  __shared__ __attribute__((aligned(32))) bf16_t ldsK[2][32 * 64];  // 8 KB
  __shared__ __attribute__((aligned(32))) bf16_t ldsV[2][64 * 32];  // 8 KB
  __shared__ __attribute__((aligned(32))) bf16_t sw8[8][16 * 32];   // 8 KB

  int wib  = threadIdx.x >> 5;
  int lane = threadIdx.x & 31;
  int tid  = threadIdx.x;
  int wave = blockIdx.x * 8 + wib;
  int bh   = wave >> 7;
  int qt   = wave & 127;
  int qt0  = (blockIdx.x * 8) & 127;
  int b_   = bh >> 4;
  int h    = bh & 15;
  bool hi  = lane >= 16;
  int  l15 = lane & 15;

  const bf16_t* kbase = Kk + (size_t)bh * SEQ * HDIM;
  const bf16_t* vbase = Vt + (size_t)bh * HDIM * SEQ;
  bf16_t* sw = sw8[wib];

  const bf16_t* qrow = Q + ((size_t)bh * SEQ + qt * 16 + l15) * HDIM;
  v16bf qa[2];
#pragma unroll
  for (int cc = 0; cc < 2; ++cc) {
    int off = cc * 32 + (hi ? 16 : 0);
    qa[cc] = make_a_frag(*(const v8bf*)(qrow + off),
                         *(const v8bf*)(qrow + off + 8));
  }

  float Mr[8], Lr[8];
#pragma unroll
  for (int r = 0; r < 8; ++r) { Mr[r] = -1e30f; Lr[r] = 0.f; }
  v8f O0 = {}, O1 = {}, O2 = {}, O3 = {};

  int nch    = qt / 2 + 1;                        // this wave's chunk count
  int nchmax = (qt0 + 7) / 2 + 1;                 // block-uniform trip count

  // prologue: stage chunk 0
  stage16(&ldsK[0][tid * 8], kbase + (size_t)tid * 8);
  stage16(&ldsV[0][tid * 8], vbase + (size_t)(tid >> 2) * SEQ + (tid & 3) * 8);
  stage_wait();
  __syncthreads();

  for (int ch = 0; ch < nchmax; ++ch) {
    int kb = ch * 32;
    const bf16_t* curK = ldsK[ch & 1];
    const bf16_t* curV = ldsV[ch & 1];
    if (ch + 1 < nchmax) {                        // prefetch next chunk
      int kb1 = kb + 32;
      stage16(&ldsK[(ch + 1) & 1][tid * 8],
              kbase + (size_t)kb1 * HDIM + tid * 8);
      stage16(&ldsV[(ch + 1) & 1][tid * 8],
              vbase + (size_t)(tid >> 2) * SEQ + kb1 + (tid & 3) * 8);
    }

    if (ch < nch) {                               // wave-uniform guard
      v16bf kf[4];                                // preload K fragments
      kf[0] = *(const v16bf*)(curK + l15 * HDIM +      (hi ? 16 : 0));
      kf[1] = *(const v16bf*)(curK + l15 * HDIM + 32 + (hi ? 16 : 0));
      kf[2] = *(const v16bf*)(curK + (16 + l15) * HDIM +      (hi ? 16 : 0));
      kf[3] = *(const v16bf*)(curK + (16 + l15) * HDIM + 32 + (hi ? 16 : 0));
      v8f s0 = {}, s1 = {};
      s0 = wmma_bf16(qa[0], kf[0], s0);
      s1 = wmma_bf16(qa[0], kf[2], s1);
      s0 = wmma_bf16(qa[1], kf[1], s0);
      s1 = wmma_bf16(qa[1], kf[3], s1);

      if (ch == nch - 1) {                        // diagonal chunk mask
#pragma unroll
        for (int r = 0; r < 8; ++r) {
          int tq = qt * 16 + r + (hi ? 8 : 0);
          s0[r] = (kb + l15      > tq) ? -1e30f : s0[r];
          s1[r] = (kb + 16 + l15 > tq) ? -1e30f : s1[r];
        }
      }

      float alpha[8];
#pragma unroll
      for (int r = 0; r < 8; ++r) {
        float mx = fmaxf(s0[r], s1[r]);
#pragma unroll
        for (int m = 1; m < 16; m <<= 1) mx = fmaxf(mx, __shfl_xor(mx, m, 32));
        float nm = fmaxf(Mr[r], mx);
        float p0 = __expf(s0[r] - nm);
        float p1 = __expf(s1[r] - nm);
        s0[r] = p0; s1[r] = p1;
        float sm = p0 + p1;
#pragma unroll
        for (int m = 1; m < 16; m <<= 1) sm += __shfl_xor(sm, m, 32);
        alpha[r] = __expf(Mr[r] - nm);
        Mr[r] = nm;
        Lr[r] = Lr[r] * alpha[r] + sm;
      }
#pragma unroll
      for (int r = 0; r < 8; ++r) {
        O0[r] *= alpha[r]; O1[r] *= alpha[r];
        O2[r] *= alpha[r]; O3[r] *= alpha[r];
      }

      // P: C-layout -> A-layout via wave-private LDS
#pragma unroll
      for (int r = 0; r < 8; ++r) {
        int m = r + (hi ? 8 : 0);
        sw[m * 32 + l15]      = (bf16_t)s0[r];
        sw[m * 32 + 16 + l15] = (bf16_t)s1[r];
      }
      asm volatile("s_wait_dscnt 0" ::: "memory");
      const bf16_t* prow = sw + l15 * 32 + (hi ? 16 : 0);
      v16bf pa = make_a_frag(*(const v8bf*)(prow), *(const v8bf*)(prow + 8));

      v16bf vf[4];                                // preload V fragments
#pragma unroll
      for (int t = 0; t < 4; ++t)
        vf[t] = *(const v16bf*)(curV + (t * 16 + l15) * 32 + (hi ? 16 : 0));
      O0 = wmma_bf16(pa, vf[0], O0);
      O1 = wmma_bf16(pa, vf[1], O1);
      O2 = wmma_bf16(pa, vf[2], O2);
      O3 = wmma_bf16(pa, vf[3], O3);
    }

    stage_wait();
    __syncthreads();
  }

  int tokbase = b_ * SEQ + qt * 16 + (hi ? 8 : 0);
#pragma unroll
  for (int r = 0; r < 8; ++r) {
    float inv = 1.f / Lr[r];
    size_t rowoff = (size_t)(tokbase + r) * EMB + h * HDIM + l15;
    Ao[rowoff +  0] = (bf16_t)(O0[r] * inv);
    Ao[rowoff + 16] = (bf16_t)(O1[r] * inv);
    Ao[rowoff + 32] = (bf16_t)(O2[r] * inv);
    Ao[rowoff + 48] = (bf16_t)(O3[r] * inv);
  }
}

// ---------------------------------------------------------------------------
// Output projection: bf16 [8192 x 1024] @ packed bf16 [1024 x 1024] + bias,
// fp32 out. Same double-buffered LDS B-panel pipeline as qkv.
// ---------------------------------------------------------------------------
__global__ void __launch_bounds__(256)
proj_kernel(const bf16_t* __restrict__ A, const bf16_t* __restrict__ Wp,
            const float* __restrict__ bias, float* __restrict__ out) {
  const int NT = EMB / 16;                        // 64 column tiles
  const int NG = NT / 8;                          // 8 n-groups
  __shared__ __attribute__((aligned(32))) bf16_t lbs[2][8 * 512]; // 16 KB

  int mg   = blockIdx.x / NG;
  int ng   = blockIdx.x - mg * NG;
  int wib  = threadIdx.x >> 5;
  int lane = threadIdx.x & 31;
  int mt   = mg * 8 + wib;
  bool hi  = lane >= 16;
  int  l15 = lane & 15;
  int  tid = threadIdx.x;

  const bf16_t* arow = A + (size_t)(mt * 16 + l15) * EMB;
  const bf16_t* bkt  = Wp + ((size_t)ng * 8) * 512 + tid * 16;
  v8f acc[8] = {};

  stage16(&lbs[0][tid * 16],     bkt);
  stage16(&lbs[0][tid * 16 + 8], bkt + 8);
  bkt += (size_t)NT * 512;
  int koff0 = hi ? 8 : 0;
  v16bf a = make_a_frag(*(const v8bf*)(arow + koff0),
                        *(const v8bf*)(arow + koff0 + 16));
  stage_wait();
  __syncthreads();

  for (int kt = 0; kt < EMB / 32; ++kt) {
    const bf16_t* cur = lbs[kt & 1];
    if (kt + 1 < EMB / 32) {
      bf16_t* nxt = lbs[(kt + 1) & 1];
      stage16(nxt + tid * 16,     bkt);
      stage16(nxt + tid * 16 + 8, bkt + 8);
      bkt += (size_t)NT * 512;
    }
    int koff = ((kt + 1 < EMB / 32) ? (kt + 1) : kt) * 32 + koff0;
    v16bf a_next = make_a_frag(*(const v8bf*)(arow + koff),
                               *(const v8bf*)(arow + koff + 16));

    v16bf bfr[8];
#pragma unroll
    for (int t = 0; t < 8; ++t)
      bfr[t] = *(const v16bf*)(cur + lane * 16 + t * 512);
#pragma unroll
    for (int t = 0; t < 8; ++t)
      acc[t] = wmma_bf16(a, bfr[t], acc[t]);
    a = a_next;

    stage_wait();
    __syncthreads();
  }

  int tokbase = mt * 16 + (hi ? 8 : 0);
#pragma unroll
  for (int t = 0; t < 8; ++t) {
    int n    = ng * 128 + t * 16 + l15;
    float bv = bias[n];
#pragma unroll
    for (int r = 0; r < 8; ++r)
      out[(size_t)(tokbase + r) * EMB + n] = acc[t][r] + bv;
  }
}

// ---------------------------------------------------------------------------
extern "C" void kernel_launch(void* const* d_in, const int* in_sizes, int n_in,
                              void* d_out, int out_size, void* d_ws, size_t ws_size,
                              hipStream_t stream) {
  const float* x      = (const float*)d_in[0];
  const float* W_attn = (const float*)d_in[1];
  const float* b_attn = (const float*)d_in[2];
  const float* W_proj = (const float*)d_in[3];
  const float* b_proj = (const float*)d_in[4];
  float* out = (float*)d_out;

  bf16_t* Q   = (bf16_t*)d_ws;
  bf16_t* Kk  = Q   + (size_t)BATCH * NHEAD * SEQ * HDIM;
  bf16_t* Vt  = Kk  + (size_t)BATCH * NHEAD * SEQ * HDIM;
  bf16_t* Ao  = Vt  + (size_t)BATCH * NHEAD * SEQ * HDIM;
  bf16_t* Xp  = Ao  + (size_t)NTOK * EMB;
  bf16_t* WaP = Xp  + (size_t)NTOK * EMB;
  bf16_t* WpP = WaP + (size_t)EMB * N3;

  pack_x_kernel<<<(NTOK * EMB) / 256, 256, 0, stream>>>(x, Xp);
  pack_w_kernel<<<(EMB * N3) / 256, 256, 0, stream>>>(W_attn, WaP, N3);
  pack_w_kernel<<<(EMB * EMB) / 256, 256, 0, stream>>>(W_proj, WpP, EMB);

  qkv_kernel<<<64 * 24, 256, 0, stream>>>(Xp, WaP, b_attn, Q, Kk, Vt);
  attn_kernel<<<(BATCH * NHEAD * (SEQ / 16)) / 8, 256, 0, stream>>>(Q, Kk, Vt,
                                                                    Ao);
  proj_kernel<<<64 * 8, 256, 0, stream>>>(Ao, WpP, b_proj, out);
}
